// CausalSelfAttention_68831145885865
// MI455X (gfx1250) — matrix operand order repaired and checked by reading, not compiled, not run on hardware
//
#include <hip/hip_runtime.h>
#include <hip/hip_bf16.h>
#include <cstdint>

typedef _Float16 v16h __attribute__((ext_vector_type(16)));
typedef _Float16 h8   __attribute__((ext_vector_type(8)));
typedef _Float16 h4   __attribute__((ext_vector_type(4)));
typedef float    v8f  __attribute__((ext_vector_type(8)));
typedef unsigned u32x4 __attribute__((ext_vector_type(4)));
typedef int      i32x4 __attribute__((ext_vector_type(4)));
typedef int      i32x8 __attribute__((ext_vector_type(8)));

#define NHEADS 16
#define HDIM   128
#define SEQ    2048
#define EMB    2048
#define BATCH  2
#define MROWS  (BATCH*SEQ)   // 4096

// ---------------------------------------------------------------- WMMA helpers
__device__ __forceinline__ v8f wmma_f16(v16h a, v16h b, v8f c) {
  return __builtin_amdgcn_wmma_f32_16x16x32_f16(false, a, false, b, (short)0, c,
                                                false, false);
}

// A-operand tile (16x32, f16): lane (l%16) = row M; elements 0-7 hold
// K = c0 + 8*hi .. +7, elements 8-15 hold K = c0 + 16 + 8*hi .. +7
__device__ __forceinline__ v16h load_atile(const _Float16* p0, int row,
                                           int stride, int c0, int hi) {
  const _Float16* p = p0 + (size_t)row * stride + c0 + hi * 8;
  union { v16h v; h8 h[2]; } u;
  u.h[0] = *(const h8*)p;
  u.h[1] = *(const h8*)(p + 16);
  return u.v;
}

// B-operand tile (32x16, f16): lane (l%16) = column N; elements 0-7 hold
// K = c0 + 8*hi .. +7, elements 8-15 hold K = c0 + 16 + 8*hi .. +7
__device__ __forceinline__ v16h load_btile(const _Float16* p0, int col,
                                           int stride, int c0, int hi) {
  const _Float16* p = p0 + (size_t)col * stride + c0 + hi * 8;
  union { v16h v; h8 h[2]; } u;
  u.h[0] = *(const h8*)p;
  u.h[1] = *(const h8*)(p + 16);
  return u.v;
}

__device__ __forceinline__ float redmax16(float v) {
#pragma unroll
  for (int m = 1; m < 16; m <<= 1) v = fmaxf(v, __shfl_xor(v, m, 32));
  return v;
}
__device__ __forceinline__ float redsum16(float v) {
#pragma unroll
  for (int m = 1; m < 16; m <<= 1) v += __shfl_xor(v, m, 32);
  return v;
}

// ------------------------------------------------ Tensor Data Mover (gfx1250)
__device__ __forceinline__ unsigned lds_offset_of(const void* p) {
  return (unsigned)(unsigned long long)(uintptr_t)p;
}

__device__ __forceinline__ void tdm_load_2d_f16(unsigned lds_off,
                                                const _Float16* gptr,
                                                unsigned tile_x, unsigned tile_y,
                                                unsigned long long row_stride_elems) {
  unsigned long long ga = (unsigned long long)(uintptr_t)gptr;
  u32x4 g0;
  g0[0] = 1u;                                         // count=1, user descriptor
  g0[1] = lds_off;                                    // lds_addr (bytes)
  g0[2] = (unsigned)(ga & 0xffffffffu);               // global_addr[31:0]
  g0[3] = (unsigned)((ga >> 32) & 0x1ffffffu) | (2u << 30);  // addr[56:32], type=2
  unsigned td0 = (unsigned)row_stride_elems;          // tensor_dim0 (elems)
  unsigned td1 = 0x100000u;                           // tensor_dim1: large, no OOB
  i32x8 g1;
  g1[0] = (int)(1u << 16);                            // wg_mask=0, data_size=1 (2B)
  g1[1] = (int)((td0 & 0xffffu) << 16);               // tensor_dim0[15:0] @ bits 63:48
  g1[2] = (int)((td0 >> 16) & 0xffffu) | (int)((td1 & 0xffffu) << 16);
  g1[3] = (int)((td1 >> 16) & 0xffffu) | (int)((tile_x & 0xffffu) << 16); // tile_dim0
  g1[4] = (int)(tile_y & 0xffffu);                    // tile_dim1; tile_dim2=0
  g1[5] = (int)(unsigned)(row_stride_elems & 0xffffffffu);     // dim0_stride[31:0]
  g1[6] = (int)(unsigned)((row_stride_elems >> 32) & 0xffffu); // dim0_stride[47:32]
  g1[7] = 0;
  i32x4 z4 = {0, 0, 0, 0};
#if __clang_major__ >= 23
  i32x8 z8 = {0, 0, 0, 0, 0, 0, 0, 0};
  __builtin_amdgcn_tensor_load_to_lds(g0, g1, z4, z4, z8, 0);
#else
  __builtin_amdgcn_tensor_load_to_lds(g0, g1, z4, z4, 0);
#endif
}

// ---------------------------------------------------------------- converters
__global__ __launch_bounds__(256) void cvt_x_f16(const float* __restrict__ x,
                                                 _Float16* __restrict__ xb) {
  int i = blockIdx.x * 256 + threadIdx.x;
  float4 f = ((const float4*)x)[i];
  h4 o; o[0] = (_Float16)f.x; o[1] = (_Float16)f.y;
  o[2] = (_Float16)f.z; o[3] = (_Float16)f.w;
  ((h4*)xb)[i] = o;
}

__global__ __launch_bounds__(256) void cvt_w_t(const float* __restrict__ W,
                                               _Float16* __restrict__ Wt) {
  __shared__ float tile[32][33];
  int bx = blockIdx.x * 32, by = blockIdx.y * 32;
  int tx = threadIdx.x, ty = threadIdx.y;
#pragma unroll
  for (int r = 0; r < 32; r += 8)
    tile[ty + r][tx] = W[(size_t)(by + ty + r) * EMB + bx + tx];
  __syncthreads();
#pragma unroll
  for (int r = 0; r < 32; r += 8)
    Wt[(size_t)(bx + ty + r) * EMB + by + tx] = (_Float16)tile[tx][ty + r];
}

// ---------------------------------------------------------------- QKV GEMM
__global__ __launch_bounds__(256) void gemm_qkv(const _Float16* __restrict__ xb,
                                                const _Float16* __restrict__ Wt_all,
                                                _Float16* __restrict__ qh,
                                                _Float16* __restrict__ kh,
                                                _Float16* __restrict__ vt) {
  __shared__ __align__(128) _Float16 lds_a[2][128 * 32];
  __shared__ __align__(128) _Float16 lds_b[2][128 * 32];
  int mat = blockIdx.z;
  const _Float16* Wt = Wt_all + (size_t)mat * EMB * EMB;
  int m0 = blockIdx.x * 128;
  int n0 = blockIdx.y * 128;
  int wave = threadIdx.x >> 5;
  int lane = threadIdx.x & 31;
  int hi = lane >> 4, lid = lane & 15;

  const int NIT = EMB / 32;
  if (wave == 0) {
    tdm_load_2d_f16(lds_offset_of(lds_a[0]), xb + (size_t)m0 * EMB, 32, 128, EMB);
    tdm_load_2d_f16(lds_offset_of(lds_b[0]), Wt + (size_t)n0 * EMB, 32, 128, EMB);
  }

  v8f acc[8] = {};
  for (int i = 0; i < NIT; i++) {
    int buf = i & 1;
    if (wave == 0) {
      if (i + 1 < NIT) {
        int e1 = (i + 1) * 32;
        tdm_load_2d_f16(lds_offset_of(lds_a[buf ^ 1]),
                        xb + (size_t)m0 * EMB + e1, 32, 128, EMB);
        tdm_load_2d_f16(lds_offset_of(lds_b[buf ^ 1]),
                        Wt + (size_t)n0 * EMB + e1, 32, 128, EMB);
        __builtin_amdgcn_s_wait_tensorcnt(2);
      } else {
        __builtin_amdgcn_s_wait_tensorcnt(0);
      }
    }
    __syncthreads();
    v16h a = load_atile(lds_a[buf], wave * 16 + lid, 32, 0, hi);
    v16h bt[8];
#pragma unroll
    for (int t = 0; t < 8; t++) bt[t] = load_btile(lds_b[buf], t * 16 + lid, 32, 0, hi);
#pragma unroll
    for (int t = 0; t < 8; t++) acc[t] = wmma_f16(a, bt[t], acc[t]);
    __syncthreads();
  }

#pragma unroll
  for (int t = 0; t < 8; t++) {
#pragma unroll
    for (int j = 0; j < 8; j++) {
      int m = m0 + wave * 16 + j + 8 * hi;
      int n = n0 + t * 16 + lid;
      _Float16 v = (_Float16)acc[t][j];
      int b = m >> 11, ls = m & (SEQ - 1);
      int h = n >> 7,  d  = n & (HDIM - 1);
      if (mat == 2)
        vt[((size_t)(b * NHEADS + h) * HDIM + d) * SEQ + ls] = v;
      else {
        _Float16* dst = (mat == 0) ? qh : kh;
        dst[((size_t)(b * NHEADS + h) * SEQ + ls) * HDIM + d] = v;
      }
    }
  }
}

// ---------------------------------------------------------------- RoPE (in place)
__global__ __launch_bounds__(256) void rope_qk(_Float16* __restrict__ qh,
                                               _Float16* __restrict__ kh) {
  int i = blockIdx.x * 256 + threadIdx.x;
  int which = i >> 22;
  int p     = i & ((1 << 22) - 1);
  int fi    = p & 63;
  int rest  = p >> 6;
  int ls    = rest & (SEQ - 1);
  _Float16* t = which ? kh : qh;
  size_t base = (size_t)rest * HDIM + fi * 2;
  float x1 = (float)t[base], x2 = (float)t[base + 1];
  float freq = __expf(-(float)fi * 0.14391156831212787f);  // ln(10000)/64
  float ang  = (float)ls * freq;
  float s, c;
  __sincosf(ang, &s, &c);
  t[base]     = (_Float16)(x1 * c - x2 * s);
  t[base + 1] = (_Float16)(x1 * s + x2 * c);
}

// ---------------------------------------------------------------- attention
// Cooperative flash attention: 8 waves share TDM-staged K (32x128) and V
// (128x32) blocks in double-buffered LDS; wave w owns q rows [q0, q0+16).
__global__ __launch_bounds__(256) void attn_fwd(const _Float16* __restrict__ qh,
                                                const _Float16* __restrict__ kh,
                                                const _Float16* __restrict__ vt,
                                                _Float16* __restrict__ ob) {
  __shared__ __align__(128) _Float16 lds_k[2][32 * 128];
  __shared__ __align__(128) _Float16 lds_v[2][128 * 32];
  __shared__ _Float16 pstage[8][16 * 32];
  int wave = threadIdx.x >> 5;
  int lane = threadIdx.x & 31;
  int hi = lane >> 4, lid = lane & 15;
  int bh = blockIdx.y;
  int q0 = blockIdx.x * 128 + wave * 16;

  const _Float16* qb = qh + (size_t)bh * SEQ * HDIM;
  const _Float16* kp = kh + (size_t)bh * SEQ * HDIM;
  const _Float16* vb = vt + (size_t)bh * HDIM * SEQ;

  v16h qa[4];
#pragma unroll
  for (int cc = 0; cc < 4; cc++) qa[cc] = load_atile(qb, q0 + lid, HDIM, cc * 32, hi);

  v8f o[8] = {};
  v8f mrow, lrow;
#pragma unroll
  for (int j = 0; j < 8; j++) { mrow[j] = -1e30f; lrow[j] = 0.0f; }

  const float scale = 0.08838834764831845f;  // 1/sqrt(128)
  const int NKB = blockIdx.x * 4 + 4;        // key blocks for the whole 128-q block

  if (wave == 0) {
    tdm_load_2d_f16(lds_offset_of(lds_k[0]), kp, 128, 32, 128);
    tdm_load_2d_f16(lds_offset_of(lds_v[0]), vb, 32, 128, SEQ);
  }

  for (int i = 0; i < NKB; i++) {
    int k0 = i * 32;
    int buf = i & 1;
    if (wave == 0) {
      if (i + 1 < NKB) {
        int k1 = k0 + 32;
        tdm_load_2d_f16(lds_offset_of(lds_k[buf ^ 1]), kp + (size_t)k1 * HDIM,
                        128, 32, 128);
        tdm_load_2d_f16(lds_offset_of(lds_v[buf ^ 1]), vb + k1, 32, 128, SEQ);
        __builtin_amdgcn_s_wait_tensorcnt(2);
      } else {
        __builtin_amdgcn_s_wait_tensorcnt(0);
      }
    }
    __syncthreads();

    if (k0 < q0 + 16) {   // causal: this wave still has live keys in this block
      const _Float16* lk = lds_k[buf];
      const _Float16* lv = lds_v[buf];
      v8f s0 = {}, s1 = {};
      v16h kt[4];
#pragma unroll
      for (int cc = 0; cc < 4; cc++) kt[cc] = load_btile(lk, lid, 128, cc * 32, hi);
#pragma unroll
      for (int cc = 0; cc < 4; cc++) s0 = wmma_f16(qa[cc], kt[cc], s0);
#pragma unroll
      for (int cc = 0; cc < 4; cc++) kt[cc] = load_btile(lk, 16 + lid, 128, cc * 32, hi);
#pragma unroll
      for (int cc = 0; cc < 4; cc++) s1 = wmma_f16(qa[cc], kt[cc], s1);

#pragma unroll
      for (int j = 0; j < 8; j++) {
        int m = q0 + j + 8 * hi;
        float a0 = s0[j] * scale, a1 = s1[j] * scale;
        if (k0 + lid > m)      a0 = -1e30f;
        if (k0 + 16 + lid > m) a1 = -1e30f;
        s0[j] = a0; s1[j] = a1;
      }
      v8f mnew, alpha;
#pragma unroll
      for (int j = 0; j < 8; j++) {
        float rm = redmax16(fmaxf(s0[j], s1[j]));
        float mn = fmaxf(mrow[j], rm);
        mnew[j]  = mn;
        alpha[j] = __expf(mrow[j] - mn);
      }
#pragma unroll
      for (int j = 0; j < 8; j++) {
        s0[j] = __expf(s0[j] - mnew[j]);
        s1[j] = __expf(s1[j] - mnew[j]);
        float rs = redsum16(s0[j] + s1[j]);
        lrow[j]  = lrow[j] * alpha[j] + rs;
        mrow[j]  = mnew[j];
      }
#pragma unroll
      for (int t = 0; t < 8; t++)
#pragma unroll
        for (int j = 0; j < 8; j++) o[t][j] *= alpha[j];

      _Float16* pb = pstage[wave];
#pragma unroll
      for (int j = 0; j < 8; j++) {
        int r = j + 8 * hi;
        pb[r * 32 + lid]      = (_Float16)s0[j];
        pb[r * 32 + 16 + lid] = (_Float16)s1[j];
      }
      asm volatile("s_wait_dscnt 0" ::: "memory");
      v16h pa = load_atile(pb, lid, 32, 0, hi);
      v16h vt4[4];
#pragma unroll
      for (int t = 0; t < 4; t++) vt4[t] = load_btile(lv, t * 16 + lid, 32, 0, hi);
#pragma unroll
      for (int t = 0; t < 4; t++) o[t] = wmma_f16(pa, vt4[t], o[t]);
#pragma unroll
      for (int t = 0; t < 4; t++) vt4[t] = load_btile(lv, (t + 4) * 16 + lid, 32, 0, hi);
#pragma unroll
      for (int t = 0; t < 4; t++) o[t + 4] = wmma_f16(pa, vt4[t], o[t + 4]);
    }
    __syncthreads();
  }

#pragma unroll
  for (int j = 0; j < 8; j++) lrow[j] = 1.0f / lrow[j];
  int b = bh >> 4, h = bh & 15;
#pragma unroll
  for (int t = 0; t < 8; t++)
#pragma unroll
    for (int j = 0; j < 8; j++) {
      int ls  = q0 + j + 8 * hi;
      int col = h * HDIM + t * 16 + lid;
      ob[(size_t)(b * SEQ + ls) * EMB + col] = (_Float16)(o[t][j] * lrow[j]);
    }
}

// ---------------------------------------------------------------- out proj
__global__ __launch_bounds__(256) void gemm_out(const _Float16* __restrict__ ab,
                                                const _Float16* __restrict__ Wot,
                                                const float* __restrict__ bo,
                                                float* __restrict__ out) {
  __shared__ __align__(128) _Float16 lds_a[2][128 * 32];
  __shared__ __align__(128) _Float16 lds_b[2][128 * 32];
  int m0 = blockIdx.x * 128;
  int n0 = blockIdx.y * 128;
  int wave = threadIdx.x >> 5;
  int lane = threadIdx.x & 31;
  int hi = lane >> 4, lid = lane & 15;

  const int NIT = EMB / 32;
  if (wave == 0) {
    tdm_load_2d_f16(lds_offset_of(lds_a[0]), ab + (size_t)m0 * EMB, 32, 128, EMB);
    tdm_load_2d_f16(lds_offset_of(lds_b[0]), Wot + (size_t)n0 * EMB, 32, 128, EMB);
  }

  v8f acc[8] = {};
  for (int i = 0; i < NIT; i++) {
    int buf = i & 1;
    if (wave == 0) {
      if (i + 1 < NIT) {
        int e1 = (i + 1) * 32;
        tdm_load_2d_f16(lds_offset_of(lds_a[buf ^ 1]),
                        ab + (size_t)m0 * EMB + e1, 32, 128, EMB);
        tdm_load_2d_f16(lds_offset_of(lds_b[buf ^ 1]),
                        Wot + (size_t)n0 * EMB + e1, 32, 128, EMB);
        __builtin_amdgcn_s_wait_tensorcnt(2);
      } else {
        __builtin_amdgcn_s_wait_tensorcnt(0);
      }
    }
    __syncthreads();
    v16h a = load_atile(lds_a[buf], wave * 16 + lid, 32, 0, hi);
    v16h bt[8];
#pragma unroll
    for (int t = 0; t < 8; t++) bt[t] = load_btile(lds_b[buf], t * 16 + lid, 32, 0, hi);
#pragma unroll
    for (int t = 0; t < 8; t++) acc[t] = wmma_f16(a, bt[t], acc[t]);
    __syncthreads();
  }

#pragma unroll
  for (int t = 0; t < 8; t++)
#pragma unroll
    for (int j = 0; j < 8; j++) {
      int m = m0 + wave * 16 + j + 8 * hi;
      int n = n0 + t * 16 + lid;
      out[(size_t)m * EMB + n] = acc[t][j] + bo[n];
    }
}

// ---------------------------------------------------------------- launcher
extern "C" void kernel_launch(void* const* d_in, const int* in_sizes, int n_in,
                              void* d_out, int out_size, void* d_ws, size_t ws_size,
                              hipStream_t stream) {
  (void)in_sizes; (void)n_in; (void)out_size; (void)ws_size;
  const float* x  = (const float*)d_in[0];
  const float* Wq = (const float*)d_in[1];
  const float* Wk = (const float*)d_in[2];
  const float* Wv = (const float*)d_in[3];
  const float* Wo = (const float*)d_in[4];
  const float* bo = (const float*)d_in[5];

  char* w = (char*)d_ws;
  const size_t SZ_XB = (size_t)MROWS * EMB * 2;
  const size_t SZ_W  = (size_t)EMB * EMB * 2;
  const size_t SZ_HD = (size_t)BATCH * NHEADS * SEQ * HDIM * 2;

  _Float16* xb     = (_Float16*)(w);                   // aliased as attn_out later
  _Float16* Wt_all = (_Float16*)(w + SZ_XB);
  _Float16* qh     = (_Float16*)(w + SZ_XB + 4 * SZ_W);
  _Float16* kh     = (_Float16*)(w + SZ_XB + 4 * SZ_W + SZ_HD);
  _Float16* vt     = (_Float16*)(w + SZ_XB + 4 * SZ_W + 2 * SZ_HD);

  cvt_x_f16<<<8192, 256, 0, stream>>>(x, xb);
  dim3 gw(64, 64), bw(32, 8);
  cvt_w_t<<<gw, bw, 0, stream>>>(Wq, Wt_all + 0 * (size_t)EMB * EMB);
  cvt_w_t<<<gw, bw, 0, stream>>>(Wk, Wt_all + 1 * (size_t)EMB * EMB);
  cvt_w_t<<<gw, bw, 0, stream>>>(Wv, Wt_all + 2 * (size_t)EMB * EMB);
  cvt_w_t<<<gw, bw, 0, stream>>>(Wo, Wt_all + 3 * (size_t)EMB * EMB);

  gemm_qkv<<<dim3(32, 16, 3), 256, 0, stream>>>(xb, Wt_all, qh, kh, vt);
  rope_qk<<<32768, 256, 0, stream>>>(qh, kh);
  attn_fwd<<<dim3(SEQ / 128, BATCH * NHEADS), 256, 0, stream>>>(qh, kh, vt, xb);
  gemm_out<<<dim3(32, 16), 256, 0, stream>>>(xb, Wt_all + 3 * (size_t)EMB * EMB, bo,
                                             (float*)d_out);
}
